// NormEMAVectorQuantizer_85435489452288
// MI455X (gfx1250) — matrix operand order repaired and checked by reading, not compiled, not run on hardware
//
#include <hip/hip_runtime.h>
#include <stdint.h>

typedef __attribute__((ext_vector_type(8)))  float  v8f;
typedef __bf16 v16bf __attribute__((ext_vector_type(16)));

#define D_DIM   512
#define WAVES   4              // waves per block in GEMM kernel
#define NT      2              // N-tiles (of 16) per K group
#define NTILE   (NT * 16)      // codebook rows per K group (32)
#define BBYTES  (NTILE * 1024) // bytes per B group (32 KB)

#if defined(__has_builtin)
#if __has_builtin(__builtin_amdgcn_sched_group_barrier)
#define SGB(mask, size, id) __builtin_amdgcn_sched_group_barrier(mask, size, id)
#endif
#endif
#ifndef SGB
#define SGB(mask, size, id)
#endif

union ABFrag {
    uint4 u4[2];
    v16bf v;
};

static __device__ __forceinline__ unsigned int f32_to_bf16(float f) {
    unsigned int u = __float_as_uint(f);
    unsigned int r = (u + 0x7FFFu + ((u >> 16) & 1u)) >> 16;  // round-to-nearest-even
    return r;
}

// CDNA5 async global->LDS copy (ASYNCcnt-tracked, no VGPR data path).
static __device__ __forceinline__ void async_copy_b128(unsigned int lds_off,
                                                       const void* gaddr) {
    asm volatile("global_load_async_to_lds_b128 %0, %1, off"
                 :: "v"(lds_off), "v"(gaddr) : "memory");
}
static __device__ __forceinline__ void wait_asynccnt0() {
    asm volatile("s_wait_asynccnt 0x0" ::: "memory");
}

// ---------------------------------------------------------------------------
// Normalize rows of [R, 512] f32 -> f32 normalized + bf16 normalized.
// One wave per row: 32 lanes x 16 floats.
// ---------------------------------------------------------------------------
__global__ __launch_bounds__(256) void vq_normalize(
    const float* __restrict__ in, float* __restrict__ outf,
    unsigned short* __restrict__ outb, int R)
{
    const int lane = threadIdx.x & 31;
    const int wave = threadIdx.x >> 5;
    const int row  = blockIdx.x * 8 + wave;
    if (row >= R) return;

    const float* src = in + (size_t)row * D_DIM;
    float4 v[4];
    float ss = 0.0f;
#pragma unroll
    for (int i = 0; i < 4; ++i) {
        v[i] = *(const float4*)(src + (i * 32 + lane) * 4);
        ss += v[i].x * v[i].x + v[i].y * v[i].y + v[i].z * v[i].z + v[i].w * v[i].w;
    }
#pragma unroll
    for (int m = 1; m < 32; m <<= 1) ss += __shfl_xor(ss, m, 32);

    const float sc = 1.0f / fmaxf(sqrtf(ss), 1e-12f);

    float* dstf = outf + (size_t)row * D_DIM;
    unsigned short* dstb = outb + (size_t)row * D_DIM;
#pragma unroll
    for (int i = 0; i < 4; ++i) {
        float4 q;
        q.x = v[i].x * sc; q.y = v[i].y * sc; q.z = v[i].z * sc; q.w = v[i].w * sc;
        const int e = (i * 32 + lane) * 4;
        *(float4*)(dstf + e) = q;
        uint2 p;
        p.x = f32_to_bf16(q.x) | (f32_to_bf16(q.y) << 16);
        p.y = f32_to_bf16(q.z) | (f32_to_bf16(q.w) << 16);
        *(uint2*)(dstb + e) = p;
    }
}

// ---------------------------------------------------------------------------
__global__ void vq_zero_loss(float* __restrict__ out, long long off) {
    if (threadIdx.x == 0 && blockIdx.x == 0) out[off] = 0.0f;
}

// ---------------------------------------------------------------------------
// sim = zf_bf16 @ en_bf16^T with running argmax over K.
// A (16 rows x 512) lives entirely in VGPRs per wave; B groups of 32 codebook
// rows are double-buffered in LDS via global_load_async_to_lds_b128 and
// shared by all 4 waves of the block. Per dstep, both sub-tile fragments are
// loaded before their two WMMAs; sched_group_barrier pins {4 DS, 2 WMMA}
// pipelining with an 8-DS prologue.
// ---------------------------------------------------------------------------
__global__ __launch_bounds__(WAVES * 32) void vq_gemm_argmax(
    const unsigned short* __restrict__ zf16,  // [N, 512] bf16 bits
    const unsigned short* __restrict__ en16,  // [K, 512] bf16 bits
    int* __restrict__ indices, int kGroups)
{
    extern __shared__ unsigned char smem[];   // 2 * BBYTES = 64 KB
    const int tid  = threadIdx.x;
    const int lane = tid & 31;
    const int wave = tid >> 5;
    const int rowBase = (blockIdx.x * WAVES + wave) * 16;
    const int half = lane >> 4;   // 0 or 1
    const int col  = lane & 15;   // N column (B/D layout) == M row (A layout)

    // ---- load this wave's A tile straight into registers (128 VGPRs) ----
    // Per ISA 16-bit A 16x32 layout: lanes 0-15 hold K 0-7 (V0-3) + 16-23
    // (V4-7); lanes 16-31 hold K 8-15 + 24-31.
    ABFrag afr[16];
    const unsigned char* abase =
        (const unsigned char*)zf16 + (size_t)(rowBase + col) * 1024 + half * 16;
#pragma unroll
    for (int d = 0; d < 16; ++d) {
        afr[d].u4[0] = *(const uint4*)(abase + d * 64);
        afr[d].u4[1] = *(const uint4*)(abase + d * 64 + 32);
    }

    // LDS byte offset of dynamic shared base (low 32 bits of the generic address)
    const unsigned int ldsBase = (unsigned int)(uintptr_t)smem;

    // ---- prologue: async-stage group 0 into buffer 0 ----
#pragma unroll
    for (int i = 0; i < BBYTES / 16 / (WAVES * 32); ++i) {  // 16 iters
        const int off = (i * (WAVES * 32) + tid) * 16;
        async_copy_b128(ldsBase + off, (const unsigned char*)en16 + off);
    }
    wait_asynccnt0();
    __syncthreads();

    float bestv[8];
    int   besti[8];
#pragma unroll
    for (int r = 0; r < 8; ++r) { bestv[r] = -3.0e38f; besti[r] = 0; }

    for (int g = 0; g < kGroups; ++g) {
        // ---- async-stage group g+1 into the other buffer ----
        if (g + 1 < kGroups) {
            const unsigned char* src =
                (const unsigned char*)en16 + (size_t)(g + 1) * BBYTES;
            const unsigned int dstBase = ldsBase + ((g + 1) & 1) * BBYTES;
            __builtin_prefetch(src + BBYTES, 0, 0);  // HBM->L2, 2 groups ahead
#pragma unroll
            for (int i = 0; i < BBYTES / 16 / (WAVES * 32); ++i) {
                const int off = (i * (WAVES * 32) + tid) * 16;
                async_copy_b128(dstBase + off, src + off);
            }
        }

        // ---- compute on buffer g&1 ----
        v8f acc[NT];
#pragma unroll
        for (int s = 0; s < NT; ++s)
#pragma unroll
            for (int j = 0; j < 8; ++j) acc[s][j] = 0.0f;

        // B 32x16 16-bit layout (matches A): lanes 0-15 = K 0-7 + 16-23,
        // lanes 16-31 = K 8-15 + 24-31, column n = lane%16.
        const unsigned char* lb =
            smem + (g & 1) * BBYTES + (size_t)col * 1024 + half * 16;

#pragma unroll
        for (int d = 0; d < 16; ++d) {
            ABFrag b0, b1;
            const unsigned char* bp = lb + d * 64;
            b0.u4[0] = *(const uint4*)(bp);
            b0.u4[1] = *(const uint4*)(bp + 32);
            b1.u4[0] = *(const uint4*)(bp + 16384);
            b1.u4[1] = *(const uint4*)(bp + 16384 + 32);
            acc[0] = __builtin_amdgcn_wmma_f32_16x16x32_bf16(
                false, afr[d].v, false, b0.v, (short)0, acc[0], false, false);
            acc[1] = __builtin_amdgcn_wmma_f32_16x16x32_bf16(
                false, afr[d].v, false, b1.v, (short)0, acc[1], false, false);
        }

        // Scheduling pipeline: 8 DS reads (2 dsteps) ahead, then per dstep
        // 2 WMMAs followed by the next dstep's 4 DS reads.
        // Totals: 8 + 14*4 = 64 DS reads, 32 WMMAs.
        SGB(0x100, 8, 0);                  // DS-read prologue (2 dsteps)
#pragma unroll
        for (int d = 0; d < 16; ++d) {
            SGB(0x008, 2, 0);              // 2 WMMAs of dstep d
            if (d < 14) SGB(0x100, 4, 0);  // 4 DS reads of dstep d+2
        }

        // ---- running argmax ----
#pragma unroll
        for (int s = 0; s < NT; ++s) {
            const int n = g * NTILE + s * 16 + col;
#pragma unroll
            for (int r = 0; r < 8; ++r) {
                const float v = acc[s][r];
                if (v > bestv[r]) { bestv[r] = v; besti[r] = n; }
            }
        }

        wait_asynccnt0();   // staging of g+1 complete
        __syncthreads();    // all waves done reading buffer g
    }

    // Reduce (max, first-index) across the 16 columns held by each half-wave.
    // D layout: VGPR r holds M=r (lanes 0-15) and M=r+8 (lanes 16-31).
#pragma unroll
    for (int r = 0; r < 8; ++r) {
#pragma unroll
        for (int m = 1; m < 16; m <<= 1) {
            const float ov = __shfl_xor(bestv[r], m, 32);
            const int   oi = __shfl_xor(besti[r], m, 32);
            if (ov > bestv[r] || (ov == bestv[r] && oi < besti[r])) {
                bestv[r] = ov; besti[r] = oi;
            }
        }
        if (col == 0) indices[rowBase + r + half * 8] = besti[r];
    }
}

// ---------------------------------------------------------------------------
// Gather winning codebook rows, write quantized output, accumulate loss,
// emit indices (as floats) into the output tail.
// ---------------------------------------------------------------------------
__global__ __launch_bounds__(256) void vq_gather_loss(
    const float* __restrict__ zf, const float* __restrict__ en,
    const int* __restrict__ indices, float* __restrict__ out,
    int N, float scale)
{
    const int lane = threadIdx.x & 31;
    const int wave = threadIdx.x >> 5;
    const int row  = blockIdx.x * 8 + wave;
    if (row >= N) return;

    const int idx = indices[row];
    const float* src = en + (size_t)idx * D_DIM;
    const float* zr  = zf + (size_t)row * D_DIM;
    float* dst = out + (size_t)row * D_DIM;

    float ss = 0.0f;
#pragma unroll
    for (int i = 0; i < 4; ++i) {
        const int e = (i * 32 + lane) * 4;
        const float4 q = *(const float4*)(src + e);
        const float4 z = *(const float4*)(zr + e);
        *(float4*)(dst + e) = q;
        const float dx = q.x - z.x, dy = q.y - z.y, dz = q.z - z.z, dw = q.w - z.w;
        ss += dx * dx + dy * dy + dz * dz + dw * dw;
    }
#pragma unroll
    for (int m = 1; m < 32; m <<= 1) ss += __shfl_xor(ss, m, 32);

    if (lane == 0) {
        atomicAdd(out + (size_t)N * D_DIM, ss * scale);          // loss slot
        out[(size_t)N * D_DIM + 1 + row] = (float)idx;           // index tail
    }
}

// ---------------------------------------------------------------------------
extern "C" void kernel_launch(void* const* d_in, const int* in_sizes, int n_in,
                              void* d_out, int out_size, void* d_ws, size_t ws_size,
                              hipStream_t stream) {
    const float* z = (const float*)d_in[0];          // [B,T,D] = [N,512]
    const float* w = (const float*)d_in[1];          // [K,512]
    float* out = (float*)d_out;

    const int N = in_sizes[0] / D_DIM;               // 16384
    const int K = in_sizes[1] / D_DIM;               // 8192

    // workspace layout
    unsigned char* ws = (unsigned char*)d_ws;
    float* en_f = (float*)ws;                               ws += (size_t)K * D_DIM * 4;
    float* zf_f = (float*)ws;                               ws += (size_t)N * D_DIM * 4;
    unsigned short* en_b = (unsigned short*)ws;             ws += (size_t)K * D_DIM * 2;
    unsigned short* zf_b = (unsigned short*)ws;             ws += (size_t)N * D_DIM * 2;
    int* idx_ws = (int*)ws;

    vq_normalize<<<K / 8, 256, 0, stream>>>(w, en_f, en_b, K);
    vq_normalize<<<N / 8, 256, 0, stream>>>(z, zf_f, zf_b, N);
    vq_zero_loss<<<1, 32, 0, stream>>>(out, (long long)N * D_DIM);

    const int kGroups = K / NTILE;                   // 256
    const int gemmBlocks = N / (WAVES * 16);         // 256
    vq_gemm_argmax<<<gemmBlocks, WAVES * 32, 2 * BBYTES, stream>>>(
        zf_b, en_b, idx_ws, kGroups);

    const float scale = 2.0f / ((float)N * (float)D_DIM);
    vq_gather_loss<<<N / 8, 256, 0, stream>>>(zf_f, en_f, idx_ws, out, N, scale);
}
